// AtomPredictor_49452253446996
// MI455X (gfx1250) — matrix operand order repaired and checked by reading, not compiled, not run on hardware
//
#include <hip/hip_runtime.h>
#include <hip/hip_bf16.h>

typedef __attribute__((ext_vector_type(16))) __bf16 v16bf;
typedef __attribute__((ext_vector_type(8)))  float  v8f;

#define D_DIM     256
#define HID_DIM   256
#define M_TILE    64    // pairs per block (4 x 16-row WMMA tiles)
#define A_STRIDE  264   // bf16 elems per LDS A row (row = 528 B, 16B-aligned)
#define P_STRIDE  17    // f32 elems per LDS partial row (pad vs 64 banks)

__device__ __forceinline__ unsigned short f32_to_bf16_rne(float f) {
    unsigned int u = __float_as_uint(f);
    u += 0x7FFFu + ((u >> 16) & 1u);   // round-to-nearest-even
    return (unsigned short)(u >> 16);
}

// Prep: WbT[n][k] = bf16(W_h[k][n]); 128 KB result stays L2-resident.
__global__ void prep_wbt(const float* __restrict__ W_h,
                         unsigned short* __restrict__ WbT) {
    int idx = blockIdx.x * blockDim.x + threadIdx.x;   // 0..65535
    int k = idx & 255;
    int n = idx >> 8;
    WbT[n * 256 + k] = f32_to_bf16_rne(W_h[k * 256 + n]);
}

__global__ void __launch_bounds__(512)
pair_mlp_kernel(const float* __restrict__ atom_h,
                const int*   __restrict__ pair_idx,
                const unsigned short* __restrict__ WbT,
                const float* __restrict__ b_h,
                const float* __restrict__ W_o,
                const float* __restrict__ b_o,
                float* __restrict__ out,
                int n_pairs)
{
    __shared__ unsigned short lds_A[M_TILE * A_STRIDE];   // 33792 B
    __shared__ float          lds_p[M_TILE * P_STRIDE];   //  4352 B

    const int tid       = threadIdx.x;
    const int pair_base = blockIdx.x * M_TILE;

    // ---------- Phase 1: gather + mean-pool + f32->bf16 into LDS A tile ----
    {
        const int mrow = tid >> 5;          // wave id: rows mrow, mrow+16, +32, +48
        const int kc   = (tid & 31) * 8;    // 8 consecutive features per lane
#pragma unroll
        for (int t = 0; t < 4; ++t) {
            int m = t * 16 + mrow;
            int p = pair_base + m;
            if (p >= n_pairs) p = n_pairs - 1;      // select, no divergence
            int i0 = pair_idx[2 * p + 0];
            int i1 = pair_idx[2 * p + 1];
            const float4* a0 = (const float4*)(atom_h + (size_t)i0 * D_DIM + kc);
            const float4* a1 = (const float4*)(atom_h + (size_t)i1 * D_DIM + kc);
            float4 x0 = a0[0], x1 = a0[1];
            float4 y0 = a1[0], y1 = a1[1];
            float v[8];
            v[0] = (x0.x + y0.x) * 0.5f;  v[1] = (x0.y + y0.y) * 0.5f;
            v[2] = (x0.z + y0.z) * 0.5f;  v[3] = (x0.w + y0.w) * 0.5f;
            v[4] = (x1.x + y1.x) * 0.5f;  v[5] = (x1.y + y1.y) * 0.5f;
            v[6] = (x1.z + y1.z) * 0.5f;  v[7] = (x1.w + y1.w) * 0.5f;
            union { unsigned short us[8]; uint4 q; } pk;
#pragma unroll
            for (int j = 0; j < 8; ++j) pk.us[j] = f32_to_bf16_rne(v[j]);
            *(uint4*)(lds_A + m * A_STRIDE + kc) = pk.q;   // one 16B LDS store
        }
    }
    __syncthreads();

    // ---------- Phase 2: 64x256 @ 256x256 via v_wmma_f32_16x16x32_bf16 -----
    const int wave   = tid >> 5;    // 0..15 -> HID column slice
    const int lane   = tid & 31;
    const int laneM  = lane & 15;
    const int laneHi = lane >> 4;
    const int n0     = wave * 16;

    v8f acc0 = {0.f,0.f,0.f,0.f,0.f,0.f,0.f,0.f};
    v8f acc1 = acc0, acc2 = acc0, acc3 = acc0;

    // B layout: K runs {kb..kb+7},{kb+16..kb+23}, kb=laneHi*8 (matches A)
    const unsigned short* Bbase = WbT + (size_t)(n0 + laneM) * D_DIM + laneHi * 8;
    // A layout: lane holds row laneM (+tile*16); K runs {kb..kb+7},{kb+16..kb+23}, kb=laneHi*8
    const unsigned short* Abase = lds_A + laneM * A_STRIDE + laneHi * 8;

#pragma unroll
    for (int kk = 0; kk < 8; ++kk) {
        const int k = kk * 32;
        union { v16bf v; uint4 q[2]; } bfrag;
        bfrag.q[0] = *(const uint4*)(Bbase + k);        // K = kB .. kB+7
        bfrag.q[1] = *(const uint4*)(Bbase + k + 16);   // K = kB+16 .. kB+23

        union { v16bf v; uint4 q[2]; } a0, a1, a2, a3;
        a0.q[0] = *(const uint4*)(Abase + k);
        a0.q[1] = *(const uint4*)(Abase + k + 16);
        a1.q[0] = *(const uint4*)(Abase + 16 * A_STRIDE + k);
        a1.q[1] = *(const uint4*)(Abase + 16 * A_STRIDE + k + 16);
        a2.q[0] = *(const uint4*)(Abase + 32 * A_STRIDE + k);
        a2.q[1] = *(const uint4*)(Abase + 32 * A_STRIDE + k + 16);
        a3.q[0] = *(const uint4*)(Abase + 48 * A_STRIDE + k);
        a3.q[1] = *(const uint4*)(Abase + 48 * A_STRIDE + k + 16);

        acc0 = __builtin_amdgcn_wmma_f32_16x16x32_bf16(false, a0.v, false, bfrag.v,
                                                       (short)0, acc0, false, false);
        acc1 = __builtin_amdgcn_wmma_f32_16x16x32_bf16(false, a1.v, false, bfrag.v,
                                                       (short)0, acc1, false, false);
        acc2 = __builtin_amdgcn_wmma_f32_16x16x32_bf16(false, a2.v, false, bfrag.v,
                                                       (short)0, acc2, false, false);
        acc3 = __builtin_amdgcn_wmma_f32_16x16x32_bf16(false, a3.v, false, bfrag.v,
                                                       (short)0, acc3, false, false);
    }

    // ---------- Phase 3: bias + ReLU + (h . W_o) partial reduce in regs ----
    // D layout: element r of acc -> M = r + 8*laneHi (+tile*16), N = laneM (+n0)
    {
        const float bias = b_h[n0 + laneM];
        const float wo   = W_o[n0 + laneM];
        v8f accs[4] = {acc0, acc1, acc2, acc3};
#pragma unroll
        for (int t = 0; t < 4; ++t) {
#pragma unroll
            for (int r = 0; r < 8; ++r) {
                float x = fmaxf(accs[t][r] + bias, 0.0f) * wo;
                // reduce over the 16-lane N group (bit-4 untouched -> stays in half)
                x += __shfl_xor(x, 1, 32);
                x += __shfl_xor(x, 2, 32);
                x += __shfl_xor(x, 4, 32);
                x += __shfl_xor(x, 8, 32);
                if (laneM == 0)
                    lds_p[(t * 16 + r + 8 * laneHi) * P_STRIDE + wave] = x;
            }
        }
    }
    __syncthreads();

    // ---------- Phase 4: sum partials across the 16 waves ------------------
    if (tid < M_TILE) {
        float s = 0.f;
#pragma unroll
        for (int w = 0; w < 16; ++w) s += lds_p[tid * P_STRIDE + w];
        int p = pair_base + tid;
        if (p < n_pairs) out[p] = s + b_o[0];
    }
}

extern "C" void kernel_launch(void* const* d_in, const int* in_sizes, int n_in,
                              void* d_out, int out_size, void* d_ws, size_t ws_size,
                              hipStream_t stream) {
    const float* atom_h   = (const float*)d_in[0];
    const int*   pair_idx = (const int*)  d_in[1];
    const float* W_h      = (const float*)d_in[2];
    const float* b_h      = (const float*)d_in[3];
    const float* W_o      = (const float*)d_in[4];
    const float* b_o      = (const float*)d_in[5];
    float*       out      = (float*)d_out;

    const int n_pairs = in_sizes[1] / 2;

    unsigned short* WbT = (unsigned short*)d_ws;   // 256*256 bf16 = 128 KB

    prep_wbt<<<256, 256, 0, stream>>>(W_h, WbT);

    const int nblocks = (n_pairs + M_TILE - 1) / M_TILE;
    pair_mlp_kernel<<<nblocks, 512, 0, stream>>>(atom_h, pair_idx, WbT,
                                                 b_h, W_o, b_o, out, n_pairs);
}
